// BiLSTM_CRF_29051158790613
// MI455X (gfx1250) — compile-verified, matrix operand order checked by reading, not run on hardware
//
#include <hip/hip_runtime.h>
#include <hip/hip_bf16.h>

// ---------------------------------------------------------------------------
// BiLSTM-CRF for MI455X (gfx1250, wave32, WMMA 16x16x32 f16)
//   T=512 B=64 EMB=512 H=512 (per dir) 4H=2048 K(tags)=16
// Pipeline: k_prep (weights fp32->fp16, zero state) -> k_embed (gather)
//        -> k_recurrent (grid=32: 16 WGs/direction, Wcat slice staged in
//           256KB LDS, ping/pong software-pipelined WMMA GEMM + LSTM
//           elementwise, cross-WG split barrier via L2 atomics)
//        -> k_emit (emissions WMMA GEMM) -> k_crf (forward algorithm)
// ---------------------------------------------------------------------------

typedef __attribute__((ext_vector_type(16))) _Float16 v16h;
typedef __attribute__((ext_vector_type(8)))  float    v8f;
typedef __attribute__((ext_vector_type(4)))  unsigned int u32x4;

#define TT   512
#define BB   64
#define EMBD 512
#define HH   512
#define FOURH 2048
#define KCAT 1024
#define KTAG 16
#define START_TAG 14
#define STOP_TAG 15
#define NWG_DIR 16            // workgroups per direction in k_recurrent
#define UNITS_WG 32           // hidden units owned per workgroup (2 N-tiles)
#define LDSW_HALFS (4 * UNITS_WG * KCAT)   // 131072 halfs = 256 KB

union F16Frag { v16h v; u32x4 q[2]; };

// Load one f16 WMMA fragment: lane holds 16 halfs as two contiguous 8-half
// (16B) groups at p and p+16 halfs. Emits 2x global_load_b128 / ds_load_b128.
__device__ __forceinline__ v16h load_frag(const _Float16* p) {
    F16Frag f;
    f.q[0] = *(const u32x4*)(p);
    f.q[1] = *(const u32x4*)(p + 16);
    return f.v;
}

__device__ __forceinline__ v8f wmma_f16(v16h a, v16h b, v8f c) {
    return __builtin_amdgcn_wmma_f32_16x16x32_f16(
        /*neg_a=*/false, a, /*neg_b=*/false, b,
        /*c_mod=*/(short)0, c, /*reuse_a=*/false, /*reuse_b=*/false);
}

__device__ __forceinline__ float sigm(float x) {
    return 1.0f / (1.0f + __expf(-x));
}
// tanh via one v_exp_f32: tanh(x) = 1 - 2/(e^{2x}+1). Inputs here are
// pre-activations of magnitude ~O(1); exp saturates gracefully at the
// extremes, matching fp16-pipeline accuracy.
__device__ __forceinline__ float fast_tanh(float x) {
    return 1.0f - 2.0f / (__expf(2.0f * x) + 1.0f);
}

// ---------------------------------------------------------------------------
// k_prep: build Wcat[dir][2048][1024] fp16 = [Wih | Whh], Wtag16[16][1024],
//         zero Hcur ping-pong state and sync counters.
// ---------------------------------------------------------------------------
#define NW ((size_t)FOURH * KCAT)              // 2,097,152 per direction
#define PREP_TOTAL (2*NW + (size_t)KTAG*KCAT + (size_t)4*BB*HH + 2)

__global__ __launch_bounds__(256)
void k_prep(const float* __restrict__ wihf, const float* __restrict__ whhf,
            const float* __restrict__ wihb, const float* __restrict__ whhb,
            const float* __restrict__ wtag,
            _Float16* __restrict__ WcatF, _Float16* __restrict__ WcatB,
            _Float16* __restrict__ Wtag16, _Float16* __restrict__ Hcur,
            unsigned int* __restrict__ syncc) {
    size_t idx = (size_t)blockIdx.x * 256 + threadIdx.x;
    if (idx < 2 * NW) {
        int dir = idx >= NW;
        size_t i = idx - (size_t)dir * NW;
        int n = (int)(i >> 10);
        int k = (int)(i & 1023);
        const float* wih = dir ? wihb : wihf;
        const float* whh = dir ? whhb : whhf;
        float v = (k < HH) ? wih[(size_t)n * EMBD + k]
                           : whh[(size_t)n * HH + (k - HH)];
        (dir ? WcatB : WcatF)[i] = (_Float16)v;
    } else if (idx < 2 * NW + (size_t)KTAG * KCAT) {
        size_t i = idx - 2 * NW;
        Wtag16[i] = (_Float16)wtag[i];
    } else if (idx < 2 * NW + (size_t)KTAG * KCAT + (size_t)4 * BB * HH) {
        Hcur[idx - (2 * NW + (size_t)KTAG * KCAT)] = (_Float16)0.0f;
    } else if (idx < PREP_TOTAL) {
        syncc[idx - (2 * NW + (size_t)KTAG * KCAT + (size_t)4 * BB * HH)] = 0u;
    }
}

// ---------------------------------------------------------------------------
// k_embed: Xemb16[t][b][e] = fp16(embed[tokens[t][b]][e])
// ---------------------------------------------------------------------------
__global__ __launch_bounds__(256)
void k_embed(const int* __restrict__ tokens, const float* __restrict__ embed,
             _Float16* __restrict__ X) {
    int tb = blockIdx.x;                       // t*B + b
    int tok = tokens[tb];
    const float* src = embed + (size_t)tok * EMBD;
    _Float16*   dst  = X + (size_t)tb * EMBD;
    for (int e = threadIdx.x; e < EMBD; e += 256)
        dst[e] = (_Float16)src[e];
}

// ---------------------------------------------------------------------------
// k_recurrent: grid = 32 (blocks 0..15 forward, 16..31 backward), 256 threads
// (8 waves). WG q owns hidden units [32q, 32q+32). Wave w: mt = w&3 (batch
// tile), utl = w>>2 (local unit tile 0..1). Per step each wave does 4 gate
// tiles x 32 K-tiles = 128 v_wmma, fused x@Wih + h@Whh. Weight slice lives in
// 256KB LDS (staged once); fragments are ping/pong double-buffered (unroll-2)
// so no register rotation movs and no full-drain waits sit between WMMAs.
// h is exchanged via an L2 ping-pong buffer with a per-direction global
// arrive/spin barrier.
// ---------------------------------------------------------------------------
__device__ __forceinline__ const _Float16* a_src(int kn, const _Float16* xrow,
                                                 const _Float16* hrow) {
    return (kn < 16) ? (xrow + kn * 32) : (hrow + (kn - 16) * 32);
}

__global__ __launch_bounds__(256, 1)
void k_recurrent(const _Float16* __restrict__ Xemb,
                 const _Float16* __restrict__ WcatF,
                 const _Float16* __restrict__ WcatB,
                 const float* __restrict__ bF, const float* __restrict__ bB,
                 _Float16* __restrict__ Hf, _Float16* __restrict__ Hb,
                 _Float16* __restrict__ Hcur, unsigned int* __restrict__ syncc) {
    extern __shared__ __align__(16) _Float16 ldsW[];   // [4][32][1024] halfs

    const int dir  = blockIdx.x >> 4;          // 0 fwd, 1 bwd
    const int q    = blockIdx.x & 15;          // unit-slice index
    const int tid  = threadIdx.x;
    const int lane = tid & 31;
    const int w    = tid >> 5;                 // wave 0..7
    const int mt   = w & 3;                    // batch tile 0..3
    const int utl  = w >> 2;                   // local unit tile 0..1
    const int Ul   = utl * 16;                 // local unit base
    const int Ug   = q * UNITS_WG + Ul;        // global unit base
    const int ln   = lane & 15;
    const int klb  = (lane < 16) ? 0 : 8;      // K-group base per lane half
    const int rbase = (lane < 16) ? 0 : 8;     // D-tile row base per lane half

    const _Float16* Wcat = dir ? WcatB : WcatF;
    const float*    bias = dir ? bB : bF;
    _Float16*       Hout = dir ? Hb : Hf;
    _Float16*       hbuf = Hcur + (size_t)dir * 2 * BB * HH;
    unsigned int*   ctr  = syncc + dir;

    // ---- stage this WG's Wcat slice into LDS (local row rl = g*32+u) ----
    for (int v = tid; v < LDSW_HALFS / 8; v += 256) {
        int rl = v >> 7;                       // 0..127 (128 16B-vecs per row)
        int cv = v & 127;
        int g = rl >> 5, u = rl & 31;
        const u32x4* src =
            (const u32x4*)(Wcat + (size_t)(g * HH + q * UNITS_WG + u) * KCAT + cv * 8);
        *(u32x4*)&ldsW[(size_t)rl * KCAT + cv * 8] = *src;
    }
    __syncthreads();

    float bg[4];
    #pragma unroll
    for (int g = 0; g < 4; ++g) bg[g] = bias[g * HH + Ug + ln];

    float cst[8];
    #pragma unroll
    for (int r = 0; r < 8; ++r) cst[r] = 0.0f;

    const int arow = mt * 16 + ln;             // A-matrix row this lane loads
    const int brl  = Ul + ln;                  // local B row base (per gate +32)

    for (int step = 0; step < TT; ++step) {
        const int t   = dir ? (TT - 1 - step) : step;
        const int par = step & 1;
        const _Float16* hin   = hbuf + (size_t)par * BB * HH;
        _Float16*       hnext = hbuf + (size_t)(1 - par) * BB * HH;
        const _Float16* xrow  = Xemb + ((size_t)t * BB + arow) * EMBD;
        const _Float16* hrow  = hin + (size_t)arow * HH;

        // Opaque zero folded into LDS offsets: stops LICM from hoisting the
        // (value-invariant) weight-fragment loads out of the step loop and
        // spilling them to scratch.
        int kz;
        asm volatile("s_mov_b32 %0, 0" : "=s"(kz));
        int lb[4];
        #pragma unroll
        for (int g = 0; g < 4; ++g)
            lb[g] = (g * UNITS_WG + brl) * KCAT + klb + kz;

        if (step + 1 < TT) {   // pull next step's activations toward the WGP
            int tn = dir ? (TT - 2 - step) : (step + 1);
            __builtin_prefetch(Xemb + ((size_t)tn * BB + arow) * EMBD, 0, 3);
        }

        v8f acc[4];
        #pragma unroll
        for (int g = 0; g < 4; ++g) {
            v8f z;
            #pragma unroll
            for (int i = 0; i < 8; ++i) z[i] = bg[g];
            acc[g] = z;
        }

        // ---- ping/pong software-pipelined GEMM ----
        // K tiles 0..15 = x@Wih, 16..31 = h@Whh; P holds even kt, Q odd kt.
        v16h aP = load_frag(xrow + klb);
        v16h bP[4];
        #pragma unroll
        for (int g = 0; g < 4; ++g) bP[g] = load_frag(&ldsW[lb[g]]);

        for (int kt = 0; kt < 32; kt += 2) {
            // load kt+1 into Q
            v16h aQ = load_frag(a_src(kt + 1, xrow, hrow) + klb);
            v16h bQ[4];
            #pragma unroll
            for (int g = 0; g < 4; ++g)
                bQ[g] = load_frag(&ldsW[lb[g] + (kt + 1) * 32]);
            // consume P (kt)
            #pragma unroll
            for (int g = 0; g < 4; ++g)
                acc[g] = wmma_f16(aP, bP[g], acc[g]);
            // load kt+2 into P
            if (kt + 2 < 32) {
                aP = load_frag(a_src(kt + 2, xrow, hrow) + klb);
                #pragma unroll
                for (int g = 0; g < 4; ++g)
                    bP[g] = load_frag(&ldsW[lb[g] + (kt + 2) * 32]);
            }
            // consume Q (kt+1)
            #pragma unroll
            for (int g = 0; g < 4; ++g)
                acc[g] = wmma_f16(aQ, bQ[g], acc[g]);
        }

        // ---- LSTM cell elementwise (gate order i,f,g,o) ----
        #pragma unroll
        for (int r = 0; r < 8; ++r) {
            float iv = acc[0][r], fv = acc[1][r], gv = acc[2][r], ov = acc[3][r];
            float c  = sigm(fv) * cst[r] + sigm(iv) * fast_tanh(gv);
            cst[r] = c;
            float h  = sigm(ov) * fast_tanh(c);
            int bidx = mt * 16 + rbase + r;
            _Float16 h16 = (_Float16)h;
            hnext[(size_t)bidx * HH + Ug + ln] = h16;
            Hout[((size_t)t * BB + bidx) * HH + Ug + ln] = h16;
        }

        // ---- per-direction cross-workgroup split barrier (16 WGs) ----
        __syncthreads();
        if (tid == 0) {
            __builtin_amdgcn_fence(__ATOMIC_RELEASE, "agent");  // global_wb
            atomicAdd(ctr, 1u);
            const unsigned int target = (unsigned)NWG_DIR * (unsigned)(step + 1);
            while (__hip_atomic_load(ctr, __ATOMIC_RELAXED,
                                     __HIP_MEMORY_SCOPE_AGENT) < target) {
                __builtin_amdgcn_s_sleep(2);
            }
            __builtin_amdgcn_fence(__ATOMIC_ACQUIRE, "agent");  // global_inv
        }
        __syncthreads();
    }
}

// ---------------------------------------------------------------------------
// k_emit: feats[m][k] = [Hf|Hb][m][:] . w_tag[k][:] + b_tag[k]
//   M = T*B = 32768 -> 2048 M-tiles; one wave per tile; 32 v_wmma per tile.
// ---------------------------------------------------------------------------
__global__ __launch_bounds__(256)
void k_emit(const _Float16* __restrict__ Hf, const _Float16* __restrict__ Hb,
            const _Float16* __restrict__ Wtag16, const float* __restrict__ btag,
            float* __restrict__ feats) {
    const int lane = threadIdx.x & 31;
    const int wv   = threadIdx.x >> 5;
    const int tile = blockIdx.x * 8 + wv;          // 0..2047
    const int ln   = lane & 15;
    const int klb  = (lane < 16) ? 0 : 8;
    const int rbase = (lane < 16) ? 0 : 8;
    const int row  = tile * 16 + ln;

    v8f acc;
    #pragma unroll
    for (int i = 0; i < 8; ++i) acc[i] = 0.0f;

    const _Float16* hf = Hf + (size_t)row * HH;
    const _Float16* hb = Hb + (size_t)row * HH;
    const _Float16* wt = Wtag16 + (size_t)ln * KCAT;
    for (int kt = 0; kt < 16; ++kt) {
        v16h a = load_frag(hf + kt * 32 + klb);
        v16h b = load_frag(wt + kt * 32 + klb);
        acc = wmma_f16(a, b, acc);
    }
    for (int kt = 0; kt < 16; ++kt) {
        v16h a = load_frag(hb + kt * 32 + klb);
        v16h b = load_frag(wt + HH + kt * 32 + klb);
        acc = wmma_f16(a, b, acc);
    }

    const float bt = btag[ln];
    #pragma unroll
    for (int r = 0; r < 8; ++r) {
        int m = tile * 16 + rbase + r;
        feats[(size_t)m * KTAG + ln] = acc[r] + bt;
    }
}

// ---------------------------------------------------------------------------
// k_crf: forward algorithm. 1024 threads = (b=0..63) x (j=0..15); alpha and
// transitions in LDS; 512 sequential steps of a 16-wide logsumexp per thread.
// ---------------------------------------------------------------------------
__global__ __launch_bounds__(1024)
void k_crf(const float* __restrict__ feats, const float* __restrict__ trans,
           float* __restrict__ out) {
    __shared__ float alpha[BB][KTAG];
    __shared__ float tr[KTAG][KTAG];
    __shared__ float term[BB][KTAG];
    const int tid = threadIdx.x;
    const int b = tid >> 4;
    const int j = tid & 15;
    if (tid < KTAG * KTAG) tr[tid >> 4][tid & 15] = trans[tid];
    alpha[b][j] = (j == START_TAG) ? 0.0f : -10000.0f;
    __syncthreads();

    for (int t = 0; t < TT; ++t) {
        float emit = feats[((size_t)t * BB + b) * KTAG + j];
        float m = -3.4e38f;
        #pragma unroll
        for (int i = 0; i < KTAG; ++i)
            m = fmaxf(m, alpha[b][i] + tr[j][i]);
        float s = 0.0f;
        #pragma unroll
        for (int i = 0; i < KTAG; ++i)
            s += __expf(alpha[b][i] + tr[j][i] - m);
        float na = m + __logf(s) + emit;
        __syncthreads();
        alpha[b][j] = na;
        __syncthreads();
    }

    term[b][j] = alpha[b][j] + tr[STOP_TAG][j];
    __syncthreads();
    if (j == 0) {
        float m = -3.4e38f;
        #pragma unroll
        for (int i = 0; i < KTAG; ++i) m = fmaxf(m, term[b][i]);
        float s = 0.0f;
        #pragma unroll
        for (int i = 0; i < KTAG; ++i) s += __expf(term[b][i] - m);
        out[b] = m + __logf(s);
    }
}

// ---------------------------------------------------------------------------
extern "C" void kernel_launch(void* const* d_in, const int* in_sizes, int n_in,
                              void* d_out, int out_size, void* d_ws, size_t ws_size,
                              hipStream_t stream) {
    const int*   tokens = (const int*)d_in[0];
    const float* embed  = (const float*)d_in[1];
    const float* wihf   = (const float*)d_in[2];
    const float* whhf   = (const float*)d_in[3];
    const float* bf     = (const float*)d_in[4];
    const float* wihb   = (const float*)d_in[5];
    const float* whhb   = (const float*)d_in[6];
    const float* bb     = (const float*)d_in[7];
    const float* wtag   = (const float*)d_in[8];
    const float* btag   = (const float*)d_in[9];
    const float* trans  = (const float*)d_in[10];
    float* out = (float*)d_out;

    // Workspace carve-up (~111.5 MB total), 256B-aligned slices.
    char* p = (char*)d_ws;
    auto carve = [&](size_t bytes) -> char* {
        char* r = p;
        p += (bytes + 255) & ~(size_t)255;
        return r;
    };
    _Float16* Xemb   = (_Float16*)carve((size_t)TT * BB * EMBD * 2);   // 33.5 MB
    _Float16* WcatF  = (_Float16*)carve((size_t)FOURH * KCAT * 2);     //  4.2 MB
    _Float16* WcatB  = (_Float16*)carve((size_t)FOURH * KCAT * 2);     //  4.2 MB
    _Float16* Wtag16 = (_Float16*)carve((size_t)KTAG * KCAT * 2);
    _Float16* Hf     = (_Float16*)carve((size_t)TT * BB * HH * 2);     // 33.5 MB
    _Float16* Hb     = (_Float16*)carve((size_t)TT * BB * HH * 2);     // 33.5 MB
    _Float16* Hcur   = (_Float16*)carve((size_t)4 * BB * HH * 2);      // ping-pong x 2 dirs
    float*    feats  = (float*)carve((size_t)TT * BB * KTAG * 4);      //  2.1 MB
    unsigned* syncc  = (unsigned*)carve(256);

    const int prep_blocks = (int)((PREP_TOTAL + 255) / 256);
    k_prep<<<prep_blocks, 256, 0, stream>>>(wihf, whhf, wihb, whhb, wtag,
                                            WcatF, WcatB, Wtag16, Hcur, syncc);
    k_embed<<<TT * BB, 256, 0, stream>>>(tokens, embed, Xemb);
    k_recurrent<<<2 * NWG_DIR, 256, (size_t)LDSW_HALFS * 2, stream>>>(
        Xemb, WcatF, WcatB, bf, bb, Hf, Hb, Hcur, syncc);
    k_emit<<<2048 / 8, 256, 0, stream>>>(Hf, Hb, Wtag16, btag, feats);
    k_crf<<<1, 1024, 0, stream>>>(feats, trans, out);
    (void)in_sizes; (void)n_in; (void)out_size; (void)ws_size;
}
